// CrysGNN_82128364634254
// MI455X (gfx1250) — matrix operand order repaired and checked by reading, not compiled, not run on hardware
//
#include <hip/hip_runtime.h>
#include <hip/hip_bf16.h>
#include <math.h>

typedef __attribute__((ext_vector_type(16))) _Float16 v16h;
typedef __attribute__((ext_vector_type(8)))  float    v8f;

#define N_ATOMS 32768
#define M_NBR   12
#define ORIG    92
#define NBR     41
#define DD      64
#define BB      512
#define NPC     64
#define NCONV   3
#define BN_EPS  1e-5f
#define TWO_D   128
#define FEAT    169   /* 2*D + NBR */
#define NEDGE   (N_ATOMS * M_NBR)   /* 393216 */

static __device__ __forceinline__ float softplusf_(float x) {
  return x > 20.f ? x : log1pf(expf(x));
}
static __device__ __forceinline__ float sigmoidf_(float x) {
  return 1.f / (1.f + expf(-x));
}

#define WMMA_F16(a, b, c) \
  __builtin_amdgcn_wmma_f32_16x16x32_f16(false, (a), false, (b), (short)0, (c), false, false)

// ---- WMMA fragment loaders (fp32 -> f16, branch-free zero padding via clamp+select) ----
// A 16x32 f16: lane L -> row (L&15); element j -> k = kbase + ((L&16)?16:0) + j
static __device__ __forceinline__ v16h frag_a(const float* __restrict__ X, int ld,
                                              int row0, int k0, int K) {
  int lane = threadIdx.x & 31;
  const float* xr = X + (size_t)(row0 + (lane & 15)) * ld;
  int kb = k0 + ((lane & 16) ? 16 : 0);
  v16h a;
#pragma unroll
  for (int j = 0; j < 16; ++j) {
    int k = kb + j;
    int kc = (k < K) ? k : (K - 1);       // clamped: load always in-bounds
    float v = xr[kc];
    a[j] = (k < K) ? (_Float16)v : (_Float16)0.0f;
  }
  return a;
}
// B 32x16 f16, W stored N x K row-major: B[k][n] = W[n*ld + k]
static __device__ __forceinline__ v16h frag_b_nk(const float* __restrict__ W, int ld,
                                                 int col0, int k0, int K, int N) {
  int lane = threadIdx.x & 31;
  int n = col0 + (lane & 15);
  bool nok = n < N;
  int nc = nok ? n : (N - 1);
  int kb = k0 + ((lane & 16) ? 16 : 0);
  const float* wr = W + (size_t)nc * ld;
  v16h b;
#pragma unroll
  for (int j = 0; j < 16; ++j) {
    int k = kb + j;
    int kc = (k < K) ? k : (K - 1);
    float v = wr[kc];
    b[j] = (nok && k < K) ? (_Float16)v : (_Float16)0.0f;
  }
  return b;
}
// B 32x16 f16, W stored K x N row-major: B[k][n] = W[k*ld + col0+n]
static __device__ __forceinline__ v16h frag_b_kn(const float* __restrict__ W, int ld,
                                                 int col0, int k0, int K, int N) {
  int lane = threadIdx.x & 31;
  int n = col0 + (lane & 15);
  bool nok = n < N;
  int nc = nok ? n : (N - 1);
  int kb = k0 + ((lane & 16) ? 16 : 0);
  v16h b;
#pragma unroll
  for (int j = 0; j < 16; ++j) {
    int k = kb + j;
    int kc = (k < K) ? k : (K - 1);
    float v = W[(size_t)kc * ld + nc];
    b[j] = (nok && k < K) ? (_Float16)v : (_Float16)0.0f;
  }
  return b;
}

// ---------------- 1) h = atom_fea @ W_emb   (32768x92 @ 92x64) ----------------
__global__ void __launch_bounds__(256) k_embed(const float* __restrict__ A,
                                               const float* __restrict__ Wemb,
                                               float* __restrict__ H) {
  int tile = blockIdx.x * 8 + (threadIdx.x >> 5);
  if (tile >= 2048 * 4) return;
  int ct = tile % 4, rt = tile / 4;
  int row0 = rt * 16, col0 = ct * 16;
  v8f acc = {};
#pragma unroll
  for (int k0 = 0; k0 < 96; k0 += 32) {
    v16h a = frag_a(A, ORIG, row0, k0, ORIG);
    v16h b = frag_b_kn(Wemb, DD, col0, k0, ORIG, DD);
    acc = WMMA_F16(a, b, acc);
  }
  int lane = threadIdx.x & 31;
  int n = lane & 15;
  int mb = (lane & 16) ? 8 : 0;
#pragma unroll
  for (int i = 0; i < 8; ++i)
    H[(size_t)(row0 + mb + i) * DD + col0 + n] = acc[i];
}

// ------- 2) Pall[n, 0:128]=h@W[:, :64]^T ; Pall[n,128:256]=h@W[:,64:128]^T -------
__global__ void __launch_bounds__(256) k_pall(const float* __restrict__ H,
                                              const float* __restrict__ Wc,
                                              float* __restrict__ Pall) {
  int tile = blockIdx.x * 8 + (threadIdx.x >> 5);
  if (tile >= 2048 * 16) return;
  int ct = tile % 16, rt = tile / 16;
  int row0 = rt * 16, col0 = ct * 16;
  const float* Wbase = Wc + ((ct >= 8) ? 64 : 0);
  int ccol = (ct >= 8) ? (col0 - 128) : col0;
  v8f acc = {};
#pragma unroll
  for (int k0 = 0; k0 < 64; k0 += 32) {
    v16h a = frag_a(H, DD, row0, k0, DD);
    v16h b = frag_b_nk(Wbase, FEAT, ccol, k0, DD, TWO_D);
    acc = WMMA_F16(a, b, acc);
  }
  int lane = threadIdx.x & 31;
  int n = lane & 15;
  int mb = (lane & 16) ? 8 : 0;
#pragma unroll
  for (int i = 0; i < 8; ++i)
    Pall[(size_t)(row0 + mb + i) * 256 + col0 + n] = acc[i];
}

// ---- 3) gated = nbr_fea@W_fea^T + P1[n] + P2[nbr] + b ; accumulate BN1 stats ----
__global__ void __launch_bounds__(256) k_gated(const float* __restrict__ NF,
                                               const int* __restrict__ IDX,
                                               const float* __restrict__ Wc,
                                               const float* __restrict__ bc,
                                               const float* __restrict__ Pall,
                                               float* __restrict__ G,
                                               float* __restrict__ sum1,
                                               float* __restrict__ sq1) {
  int tile = blockIdx.x * 8 + (threadIdx.x >> 5);
  const int MT = NEDGE / 16;            // 24576
  if (tile >= MT * 8) return;
  int ct = tile % 8, rt = tile / 8;
  int row0 = rt * 16, col0 = ct * 16;
  v8f acc = {};
#pragma unroll
  for (int k0 = 0; k0 < 64; k0 += 32) {
    v16h a = frag_a(NF, NBR, row0, k0, NBR);
    v16h b = frag_b_nk(Wc + TWO_D, FEAT, col0, k0, NBR, TWO_D);
    acc = WMMA_F16(a, b, acc);
  }
  int lane = threadIdx.x & 31;
  int n = lane & 15;
  int mb = (lane & 16) ? 8 : 0;
  int c = col0 + n;
  float bias = bc[c];
  float ls = 0.f, lq = 0.f;
#pragma unroll
  for (int i = 0; i < 8; ++i) {
    int r = row0 + mb + i;              // edge row = n*12 + m
    int na = r / M_NBR;
    float g = acc[i] + Pall[(size_t)na * 256 + c]
                     + Pall[(size_t)IDX[r] * 256 + 128 + c] + bias;
    G[(size_t)r * TWO_D + c] = g;
    ls += g; lq += g * g;
  }
  ls += __shfl_xor(ls, 16, 32);
  lq += __shfl_xor(lq, 16, 32);
  if (lane < 16) {
    atomicAdd(&sum1[c], ls);
    atomicAdd(&sq1[c], lq);
  }
}

// ---- BN finalize: scale = g*rsqrt(var+eps), shift = b - mean*scale ----
__global__ void k_finalize(const float* __restrict__ sum, const float* __restrict__ sq,
                           const float* __restrict__ g, const float* __restrict__ b,
                           float* __restrict__ scale, float* __restrict__ shift,
                           float inv_count, int C) {
  int c = blockIdx.x * blockDim.x + threadIdx.x;
  if (c >= C) return;
  float m = sum[c] * inv_count;
  float v = sq[c] * inv_count - m * m;
  float inv = rsqrtf(v + BN_EPS);
  float s = inv * g[c];
  scale[c] = s;
  shift[c] = b[c] - m * s;
}

// ---- 4) summed[n,c] = sum_m sigmoid(bn(filt)) * softplus(bn(core)); BN2 stats ----
__global__ void __launch_bounds__(256) k_pool_nbr(const float* __restrict__ G,
                                                  const float* __restrict__ scale1,
                                                  const float* __restrict__ shift1,
                                                  float* __restrict__ S,
                                                  float* __restrict__ sum2,
                                                  float* __restrict__ sq2) {
  __shared__ float bs[64], bq[64];
  int t = threadIdx.x;
  if (t < 64) { bs[t] = 0.f; bq[t] = 0.f; }
  __syncthreads();
  int id = blockIdx.x * 256 + t;        // exactly 32768*64 threads
  int n = id >> 6, c = id & 63;
  float scf = scale1[c],      shf = shift1[c];
  float scs = scale1[64 + c], shs = shift1[64 + c];
  const float* gb = G + (size_t)n * M_NBR * TWO_D;
  float acc = 0.f;
#pragma unroll
  for (int m = 0; m < M_NBR; ++m) {
    float f = gb[m * TWO_D + c]      * scf + shf;
    float s = gb[m * TWO_D + 64 + c] * scs + shs;
    acc += sigmoidf_(f) * softplusf_(s);
  }
  S[id] = acc;
  atomicAdd(&bs[c], acc);
  atomicAdd(&bq[c], acc * acc);
  __syncthreads();
  if (t < 64) { atomicAdd(&sum2[t], bs[t]); atomicAdd(&sq2[t], bq[t]); }
}

// ---- 5) h = softplus(h + bn2(summed)) ----
__global__ void __launch_bounds__(256) k_update_h(float* __restrict__ H,
                                                  const float* __restrict__ S,
                                                  const float* __restrict__ scale2,
                                                  const float* __restrict__ shift2) {
  int id = blockIdx.x * 256 + threadIdx.x;
  int c = id & 63;
  H[id] = softplusf_(H[id] + S[id] * scale2[c] + shift2[c]);
}

// ---- 6) per-crystal: emb = h/||h||, z_G = mean(emb) ----
__global__ void __launch_bounds__(256) k_crystal(const float* __restrict__ H,
                                                 const int* __restrict__ CIA,
                                                 float* __restrict__ EMB,
                                                 float* __restrict__ ZG) {
  __shared__ float norms[64];
  __shared__ float tile[64 * 64];
  int b = blockIdx.x, t = threadIdx.x;
  if (t < 64) {
    int a = CIA[b * 64 + t];
    const float* hr = H + (size_t)a * 64;
    float s = 0.f;
    for (int d = 0; d < 64; ++d) { float v = hr[d]; s += v * v; }
    norms[t] = fmaxf(sqrtf(s), 1e-12f);
  }
  __syncthreads();
#pragma unroll
  for (int it = 0; it < 16; ++it) {
    int e = t + it * 256;
    int j = e >> 6, d = e & 63;
    int a = CIA[b * 64 + j];
    float v = H[(size_t)a * 64 + d] / norms[j];
    tile[e] = v;
    EMB[(size_t)b * 4096 + e] = v;
  }
  __syncthreads();
  if (t < 64) {
    float s = 0.f;
    for (int j = 0; j < 64; ++j) s += tile[j * 64 + t];
    ZG[b * 64 + t] = s * (1.f / 64.f);
  }
}

// ---- 7) t[b,n,k,e] = sum_d emb[b,n,d] * W_bil[k,d,e] ----
__global__ void __launch_bounds__(256) k_bil_t(const float* __restrict__ EMB,
                                               const float* __restrict__ Wbil,
                                               float* __restrict__ T) {
  int tile = blockIdx.x * 8 + (threadIdx.x >> 5);
  if (tile >= 2048 * 24) return;
  int ct = tile % 24, rt = tile / 24;
  int kb_ = ct >> 2;
  int col0 = (ct & 3) * 16;
  int row0 = rt * 16;
  const float* Wk = Wbil + (size_t)kb_ * 64 * 64;
  v8f acc = {};
#pragma unroll
  for (int k0 = 0; k0 < 64; k0 += 32) {
    v16h a = frag_a(EMB, 64, row0, k0, 64);
    v16h b = frag_b_kn(Wk, 64, col0, k0, 64, 64);
    acc = WMMA_F16(a, b, acc);
  }
  int lane = threadIdx.x & 31;
  int n = lane & 15;
  int mb = (lane & 16) ? 8 : 0;
#pragma unroll
  for (int i = 0; i < 8; ++i)
    T[(size_t)(row0 + mb + i) * 384 + kb_ * 64 + col0 + n] = acc[i];
}

// ---- 8) ep[b,n,m,k] = sum_e t[b,n,k,e]*emb[b,m,e] + b_bil[k] ----
__global__ void __launch_bounds__(256) k_bil_ep(const float* __restrict__ T,
                                                const float* __restrict__ EMB,
                                                const float* __restrict__ bbil,
                                                float* __restrict__ EP) {
  int tile = blockIdx.x * 8 + (threadIdx.x >> 5);
  if (tile >= 512 * 6 * 16) return;
  int ct = tile & 3; int tmp = tile >> 2;
  int rt = tmp & 3;  tmp >>= 2;
  int k = tmp % 6;   int b = tmp / 6;
  int row0 = rt * 16, col0 = ct * 16;
  const float* Ab = T + (size_t)b * 64 * 384 + k * 64;
  const float* Bb = EMB + (size_t)b * 4096;
  v8f acc = {};
#pragma unroll
  for (int k0 = 0; k0 < 64; k0 += 32) {
    v16h a = frag_a(Ab, 384, row0, k0, 64);
    v16h bf = frag_b_nk(Bb, 64, col0, k0, 64, 64);
    acc = WMMA_F16(a, bf, acc);
  }
  int lane = threadIdx.x & 31;
  int n = lane & 15;
  int mb = (lane & 16) ? 8 : 0;
  float bias = bbil[k];
#pragma unroll
  for (int i = 0; i < 8; ++i) {
    int nn = row0 + mb + i;
    int mm = col0 + n;
    EP[(((size_t)b * 64 + nn) * 64 + mm) * 6 + k] = acc[i] + bias;
  }
}

// ---- 9) edge_prob = log_softmax(ep @ W1^T + b1) over k=6 ----
__global__ void __launch_bounds__(256) k_edge(const float* __restrict__ EP,
                                              const float* __restrict__ W1,
                                              const float* __restrict__ b1,
                                              float* __restrict__ OUT) {
  int id = blockIdx.x * 256 + threadIdx.x;    // 2097152 threads
  const float* e = EP + (size_t)id * 6;
  float e0[6], e1[6];
#pragma unroll
  for (int k = 0; k < 6; ++k) e0[k] = e[k];
  float mx = -__builtin_inff();
#pragma unroll
  for (int i = 0; i < 6; ++i) {
    float s = b1[i];
#pragma unroll
    for (int k = 0; k < 6; ++k) s += e0[k] * W1[i * 6 + k];
    e1[i] = s;
    mx = fmaxf(mx, s);
  }
  float se = 0.f;
#pragma unroll
  for (int i = 0; i < 6; ++i) se += expf(e1[i] - mx);
  float lse = mx + logf(se);
  float* o = OUT + (size_t)id * 6;
#pragma unroll
  for (int i = 0; i < 6; ++i) o[i] = e1[i] - lse;
}

// ---- 10) atom_feature = emb @ W_af^T + b_af  (32768x64 @ 64x92) ----
__global__ void __launch_bounds__(256) k_atomfeat(const float* __restrict__ EMB,
                                                  const float* __restrict__ Waf,
                                                  const float* __restrict__ baf,
                                                  float* __restrict__ OUT) {
  int tile = blockIdx.x * 8 + (threadIdx.x >> 5);
  if (tile >= 2048 * 6) return;
  int ct = tile % 6, rt = tile / 6;
  int row0 = rt * 16, col0 = ct * 16;
  v8f acc = {};
#pragma unroll
  for (int k0 = 0; k0 < 64; k0 += 32) {
    v16h a = frag_a(EMB, 64, row0, k0, 64);
    v16h b = frag_b_nk(Waf, 64, col0, k0, 64, ORIG);
    acc = WMMA_F16(a, b, acc);
  }
  int lane = threadIdx.x & 31;
  int n = lane & 15;
  int mb = (lane & 16) ? 8 : 0;
  int c = col0 + n;
  if (c < ORIG) {
    float bias = baf[c];
#pragma unroll
    for (int i = 0; i < 8; ++i)
      OUT[(size_t)(row0 + mb + i) * ORIG + c] = acc[i] + bias;
  }
}

// ---- 11) sg_pred = log_softmax(z_G @ W_sg^T + b_sg) over 230 ----
__global__ void __launch_bounds__(256) k_sg(const float* __restrict__ ZG,
                                            const float* __restrict__ Wsg,
                                            const float* __restrict__ bsg,
                                            float* __restrict__ OUT) {
  __shared__ float z[64];
  __shared__ float logit[256];
  __shared__ float red[256];
  int b = blockIdx.x, t = threadIdx.x;
  if (t < 64) z[t] = ZG[b * 64 + t];
  __syncthreads();
  float v = -__builtin_inff();
  if (t < 230) {
    float s = bsg[t];
    const float* wr = Wsg + (size_t)t * 64;
    for (int d = 0; d < 64; ++d) s += z[d] * wr[d];
    v = s;
  }
  logit[t] = v;
  red[t] = v;
  __syncthreads();
  for (int off = 128; off > 0; off >>= 1) {
    if (t < off) red[t] = fmaxf(red[t], red[t + off]);
    __syncthreads();
  }
  float mx = red[0];
  __syncthreads();
  red[t] = (t < 230) ? expf(logit[t] - mx) : 0.f;
  __syncthreads();
  for (int off = 128; off > 0; off >>= 1) {
    if (t < off) red[t] += red[t + off];
    __syncthreads();
  }
  float lse = mx + logf(red[0]);
  if (t < 230) OUT[b * 230 + t] = logit[t] - lse;
}

extern "C" void kernel_launch(void* const* d_in, const int* in_sizes, int n_in,
                              void* d_out, int out_size, void* d_ws, size_t ws_size,
                              hipStream_t stream) {
  (void)in_sizes; (void)n_in; (void)out_size; (void)ws_size;
  const float* atom_fea = (const float*)d_in[0];
  const float* nbr_fea  = (const float*)d_in[1];
  const int*   nbr_idx  = (const int*)d_in[2];
  const int*   cia      = (const int*)d_in[3];
  const float* W_emb    = (const float*)d_in[4];
  const float* conv_W   = (const float*)d_in[5];
  const float* conv_b   = (const float*)d_in[6];
  const float* bn1_g    = (const float*)d_in[7];
  const float* bn1_b    = (const float*)d_in[8];
  const float* bn2_g    = (const float*)d_in[9];
  const float* bn2_b    = (const float*)d_in[10];
  const float* W_bil    = (const float*)d_in[11];
  const float* b_bil    = (const float*)d_in[12];
  const float* W1       = (const float*)d_in[13];
  const float* b1       = (const float*)d_in[14];
  const float* W_af     = (const float*)d_in[15];
  const float* b_af     = (const float*)d_in[16];
  const float* W_sg     = (const float*)d_in[17];
  const float* b_sg     = (const float*)d_in[18];

  float* ws    = (float*)d_ws;
  float* H     = ws;                  // 2,097,152
  float* Pall  = H + 2097152;         // 8,388,608
  float* S     = Pall + 8388608;      // 2,097,152
  float* stats = S + 2097152;         // 1,024
  float* big   = stats + 1024;        // 50,331,648 (G; reused for T/EP)
  float* G  = big;
  float* T  = big;
  float* EP = big + 12582912;

  float* sum1 = stats,        *sq1   = stats + 128;
  float* scale1 = stats + 256, *shift1 = stats + 384;
  float* sum2 = stats + 512,  *sq2   = stats + 576;
  float* scale2 = stats + 640, *shift2 = stats + 704;

  float* out      = (float*)d_out;
  float* out_edge = out;                       // 512*64*64*6 = 12,582,912
  float* out_af   = out + 12582912;            // 32768*92    =  3,014,656
  float* out_sg   = out_af + 3014656;          // 512*230     =    117,760
  float* out_zg   = out_sg + 117760;           // 512*64      =     32,768
  float* out_emb  = out_zg + 32768;            // 512*64*64   =  2,097,152

  k_embed<<<1024, 256, 0, stream>>>(atom_fea, W_emb, H);

  for (int i = 0; i < NCONV; ++i) {
    const float* Wc = conv_W + (size_t)i * TWO_D * FEAT;
    const float* bc = conv_b + (size_t)i * TWO_D;
    hipMemsetAsync(stats, 0, 1024 * sizeof(float), stream);
    k_pall<<<4096, 256, 0, stream>>>(H, Wc, Pall);
    k_gated<<<24576, 256, 0, stream>>>(nbr_fea, nbr_idx, Wc, bc, Pall, G, sum1, sq1);
    k_finalize<<<1, 128, 0, stream>>>(sum1, sq1, bn1_g + i * 128, bn1_b + i * 128,
                                      scale1, shift1, 1.f / (float)NEDGE, 128);
    k_pool_nbr<<<8192, 256, 0, stream>>>(G, scale1, shift1, S, sum2, sq2);
    k_finalize<<<1, 64, 0, stream>>>(sum2, sq2, bn2_g + i * 64, bn2_b + i * 64,
                                     scale2, shift2, 1.f / (float)N_ATOMS, 64);
    k_update_h<<<8192, 256, 0, stream>>>(H, S, scale2, shift2);
  }

  k_crystal<<<512, 256, 0, stream>>>(H, cia, out_emb, out_zg);
  k_bil_t<<<6144, 256, 0, stream>>>(out_emb, W_bil, T);
  k_bil_ep<<<6144, 256, 0, stream>>>(T, out_emb, b_bil, EP);
  k_edge<<<8192, 256, 0, stream>>>(EP, W1, b1, out_edge);
  k_atomfeat<<<1536, 256, 0, stream>>>(out_emb, W_af, b_af, out_af);
  k_sg<<<512, 256, 0, stream>>>(out_zg, W_sg, b_sg, out_sg);
}
